// NodeGCN_82910048682339
// MI455X (gfx1250) — matrix operand order repaired and checked by reading, not compiled, not run on hardware
//
#include <hip/hip_runtime.h>
#include <hip/hip_bf16.h>
#include <math.h>

typedef __attribute__((ext_vector_type(16))) _Float16 v16h;
typedef __attribute__((ext_vector_type(8)))  float    v8f;

#define FDIM 128
#define COUT 40
#define CPAD 48

// Involutive permutation of K within each 32-block: swap octets 1 and 2.
// Fragment layout wants lanes 0-15 holding K {0-7,16-23}, lanes 16-31 {8-15,24-31};
// after this permutation each lane's 16 halfs are CONTIGUOUS in memory.
__device__ __forceinline__ int kperm(int k) {
  int sw = (((k >> 3) & 1) ^ ((k >> 4) & 1)) ? 24 : 0;
  return k ^ sw;
}

__device__ __forceinline__ unsigned mix32(unsigned x) {
  x ^= x >> 16; x *= 0x7feb352dU;
  x ^= x >> 15; x *= 0x846ca68bU;
  x ^= x >> 16; return x;
}

// ---------------- small elementwise kernels ----------------
__global__ void k_fill1(float* __restrict__ p, int n) {
  int i = blockIdx.x * blockDim.x + threadIdx.x;
  if (i < n) p[i] = 1.0f;                     // self-loop contributes 1 to degree
}

__global__ void k_deg_acc(float* __restrict__ deg, const int* __restrict__ dst, int E) {
  int e = blockIdx.x * blockDim.x + threadIdx.x;
  if (e < E) atomicAdd(&deg[dst[e]], 1.0f);
}

__global__ void k_rsqrt_inplace(float* __restrict__ p, int n) {
  int i = blockIdx.x * blockDim.x + threadIdx.x;
  if (i < n) { float d = p[i]; p[i] = d > 0.0f ? rsqrtf(d) : 0.0f; }
}

// f32 [N][128] row-major -> f16 with per-32 K-octet swap (A-fragment-ready layout)
__global__ void k_cvt_perm(_Float16* __restrict__ o, const float* __restrict__ s, int n) {
  int i = blockIdx.x * blockDim.x + threadIdx.x;
  if (i < n) {
    int k = i & (FDIM - 1);
    int src = (i & ~(FDIM - 1)) | kperm(k);   // perm is an involution
    o[i] = (_Float16)s[src];
  }
}

// Pack weight f32 [128][ncols] row-major into per-lane WMMA B fragments:
// o[((t*ntiles + u)*32 + lane)*16 + j] = W[(32t + kperm(16*(lane>>4) + j))*ncols + 16u + (lane&15)]
// (zero-fill for padded columns)
__global__ void k_pack_w(_Float16* __restrict__ o, const float* __restrict__ w,
                         int ntiles, int ncols) {
  int i = blockIdx.x * blockDim.x + threadIdx.x;
  int total = 4 * ntiles * 32 * 16;
  if (i < total) {
    int j    = i & 15;
    int lane = (i >> 4) & 31;
    int tile = i >> 9;                 // t*ntiles + u
    int u = tile % ntiles;
    int t = tile / ntiles;
    int n = (u << 4) + (lane & 15);
    int k = (t << 5) + kperm(((lane >> 4) << 4) + j);
    o[i] = (n < ncols) ? (_Float16)w[k * ncols + n] : (_Float16)0.0f;
  }
}

// acc[node*F+f] = bias[f] + h[node*ldh+f] * dinv[node]^2   (bias + self-loop term)
__global__ void k_agg_init(float* __restrict__ acc, const float* __restrict__ h,
                           const float* __restrict__ bias, const float* __restrict__ dinv,
                           int N, int F, int ldh) {
  int i = blockIdx.x * blockDim.x + threadIdx.x;
  if (i < N * F) {
    int node = i / F, f = i - node * F;
    float di = dinv[node];
    acc[i] = bias[f] + h[(size_t)node * ldh + f] * di * di;
  }
}

// one wave per edge: acc[dst*F+f] += h[src*ldh+f] * dinv[src]*dinv[dst]
__global__ void k_agg_edges(float* __restrict__ acc, const float* __restrict__ h,
                            const float* __restrict__ dinv,
                            const int* __restrict__ src, const int* __restrict__ dst,
                            int E, int F, int ldh) {
  int lane   = threadIdx.x & 31;
  int wave   = (blockIdx.x * blockDim.x + threadIdx.x) >> 5;
  int nwaves = (gridDim.x * blockDim.x) >> 5;
  for (int e = wave; e < E; e += nwaves) {
    int s = src[e], d = dst[e];
    float nrm = dinv[s] * dinv[d];
    const float* hs = h + (size_t)s * ldh;
    float* ad = acc + (size_t)d * F;
    for (int f = lane; f < F; f += 32) atomicAdd(&ad[f], hs[f] * nrm);
  }
}

// ELU + dropout(p=0.5, deterministic hash) -> f16 in A-fragment (K-octet-swapped) layout
__global__ void k_act_drop(_Float16* __restrict__ o, const float* __restrict__ in,
                           int n, unsigned seed) {
  int i = blockIdx.x * blockDim.x + threadIdx.x;
  if (i < n) {
    int k = i & (FDIM - 1);
    int src = (i & ~(FDIM - 1)) | kperm(k);
    float v = in[src];
    v = (v > 0.0f) ? v : (expf(v) - 1.0f);               // ELU, alpha=1
    unsigned hsh = mix32((unsigned)src ^ seed);
    v = (hsh & 1u) ? v * 2.0f : 0.0f;                    // keep-prob 0.5, scaled
    o[i] = (_Float16)v;
  }
}

// ---------------- WMMA GEMM ----------------
// A: f16 [Nrows][128], K-octet-swapped rows (lane's 16 halfs contiguous).
// Bp: pre-packed B fragments. One wave computes one 16x16 tile; wave w owns cols [16w,16w+16).
// K=128 -> 4 chained v_wmma_f32_16x16x32_f16; all loads are 32B v16h (2x b128).
// NTILES/LDO are compile-time so B indexing and store offsets fold to immediates.
template<int NTILES, int LDO>
__global__ void k_gemm_wmma(const _Float16* __restrict__ A, const _Float16* __restrict__ Bp,
                            float* __restrict__ Out, int Nrows) {
  const int lane = threadIdx.x & 31;
  const int wave = threadIdx.x >> 5;
  const int m0 = blockIdx.x << 4;
  const int hi = lane >> 4;
  int row = m0 + (lane & 15);
  if (row >= Nrows) row = Nrows - 1;     // clamp loads; stores guarded below
  const v16h* __restrict__ arow = (const v16h*)(A + (size_t)row * FDIM);
  const v16h* __restrict__ bfr  = (const v16h*)Bp + (size_t)wave * 32 + lane;

  v8f c = {};
  #pragma unroll
  for (int t = 0; t < 4; ++t) {
    v16h a = arow[t * 2 + hi];
    v16h b = bfr[t * NTILES * 32];
    c = __builtin_amdgcn_wmma_f32_16x16x32_f16(false, a, false, b, (short)0, c, false, false);
  }

  // C/D layout: VGPR r -> row m0 + r + 8*hi, col 16*wave + (lane&15)
  const int n = (wave << 4) + (lane & 15);
  const int rbase = m0 + hi * 8;
  float* __restrict__ op = Out + (size_t)rbase * LDO + n;
  if (m0 + 16 <= Nrows) {                // wave-uniform: full tile -> straight-line stores
    #pragma unroll
    for (int r = 0; r < 8; ++r) op[r * LDO] = c[r];
  } else {                               // tail tile (absent when Nrows % 16 == 0)
    #pragma unroll
    for (int r = 0; r < 8; ++r) if (rbase + r < Nrows) op[r * LDO] = c[r];
  }
}

// ---------------- log-softmax over 40 classes, one wave per node ----------------
__global__ void k_log_softmax40(float* __restrict__ out, int N) {
  int lane = threadIdx.x & 31;
  int node = (blockIdx.x * blockDim.x + threadIdx.x) >> 5;
  if (node >= N) return;
  float* row = out + (size_t)node * COUT;
  float v0 = row[lane];
  float v1 = (lane < COUT - 32) ? row[32 + lane] : -INFINITY;
  float m = fmaxf(v0, v1);
  #pragma unroll
  for (int o = 16; o > 0; o >>= 1) m = fmaxf(m, __shfl_xor(m, o, 32));
  float s = expf(v0 - m) + ((lane < COUT - 32) ? expf(v1 - m) : 0.0f);
  #pragma unroll
  for (int o = 16; o > 0; o >>= 1) s += __shfl_xor(s, o, 32);
  float lg = m + logf(s);
  row[lane] = v0 - lg;
  if (lane < COUT - 32) row[32 + lane] = v1 - lg;
}

// ---------------- driver ----------------
static inline size_t alignup(size_t x) { return (x + 255) & ~(size_t)255; }

extern "C" void kernel_launch(void* const* d_in, const int* in_sizes, int n_in,
                              void* d_out, int out_size, void* d_ws, size_t ws_size,
                              hipStream_t stream) {
  const float* x  = (const float*)d_in[0];
  const int*   ei = (const int*)  d_in[1];
  const float* W1 = (const float*)d_in[2];
  const float* b1 = (const float*)d_in[3];
  const float* W2 = (const float*)d_in[4];
  const float* b2 = (const float*)d_in[5];
  const float* W3 = (const float*)d_in[6];
  const float* b3 = (const float*)d_in[7];
  float* out = (float*)d_out;

  const int N = in_sizes[0] / FDIM;        // 100000
  const int E = in_sizes[1] / 2;           // 1600000
  const int* esrc = ei;
  const int* edst = ei + E;

  // ---- workspace carve (~128 MB) ----
  char* ws = (char*)d_ws;
  float*    dinv  = (float*)ws;    ws += alignup((size_t)N * 4);
  _Float16* hf16  = (_Float16*)ws; ws += alignup((size_t)N * FDIM * 2);
  float*    hgemm = (float*)ws;    ws += alignup((size_t)N * FDIM * 4);
  float*    hagg  = (float*)ws;    ws += alignup((size_t)N * FDIM * 4);
  _Float16* W1p   = (_Float16*)ws; ws += alignup((size_t)4 * 8 * 32 * 16 * 2);
  _Float16* W2p   = (_Float16*)ws; ws += alignup((size_t)4 * 8 * 32 * 16 * 2);
  _Float16* W3p   = (_Float16*)ws; ws += alignup((size_t)4 * 3 * 32 * 16 * 2);
  float*    hg3   = hgemm;                 // alias: layer-2 GEMM output dead by then

  const int NF = N * FDIM;
  const int TB = 256;
  dim3 blk(TB);

  // degrees -> dinv (with self-loop)
  k_fill1        <<<(N + TB - 1) / TB, blk, 0, stream>>>(dinv, N);
  k_deg_acc      <<<(E + TB - 1) / TB, blk, 0, stream>>>(dinv, edst, E);
  k_rsqrt_inplace<<<(N + TB - 1) / TB, blk, 0, stream>>>(dinv, N);

  // f16 conversion (fragment-ready layouts)
  k_cvt_perm<<<(NF + TB - 1) / TB, blk, 0, stream>>>(hf16, x, NF);
  k_pack_w  <<<(4 * 8 * 512 + TB - 1) / TB, blk, 0, stream>>>(W1p, W1, 8, FDIM);
  k_pack_w  <<<(4 * 8 * 512 + TB - 1) / TB, blk, 0, stream>>>(W2p, W2, 8, FDIM);
  k_pack_w  <<<(4 * 3 * 512 + TB - 1) / TB, blk, 0, stream>>>(W3p, W3, 3, COUT);

  const int mtiles  = (N + 15) / 16;
  const int eblocks = (E + 7) / 8;         // 8 waves/block, one wave per edge

  // ---- layer 1 ----
  k_gemm_wmma<8, FDIM><<<mtiles, 256, 0, stream>>>(hf16, W1p, hgemm, N);
  k_agg_init <<<(NF + TB - 1) / TB, blk, 0, stream>>>(hagg, hgemm, b1, dinv, N, FDIM, FDIM);
  k_agg_edges<<<eblocks, blk, 0, stream>>>(hagg, hgemm, dinv, esrc, edst, E, FDIM, FDIM);
  k_act_drop <<<(NF + TB - 1) / TB, blk, 0, stream>>>(hf16, hagg, NF, 0xC0FFEE01u);

  // ---- layer 2 ----
  k_gemm_wmma<8, FDIM><<<mtiles, 256, 0, stream>>>(hf16, W2p, hgemm, N);
  k_agg_init <<<(NF + TB - 1) / TB, blk, 0, stream>>>(hagg, hgemm, b2, dinv, N, FDIM, FDIM);
  k_agg_edges<<<eblocks, blk, 0, stream>>>(hagg, hgemm, dinv, esrc, edst, E, FDIM, FDIM);
  k_act_drop <<<(NF + TB - 1) / TB, blk, 0, stream>>>(hf16, hagg, NF, 0xC0FFEE02u);

  // ---- layer 3 (40 cols padded to 48 -> 3 WMMA column tiles) ----
  k_gemm_wmma<3, CPAD><<<mtiles, 96, 0, stream>>>(hf16, W3p, hg3, N);
  k_agg_init <<<(N * COUT + TB - 1) / TB, blk, 0, stream>>>(out, hg3, b3, dinv, N, COUT, CPAD);
  k_agg_edges<<<eblocks, blk, 0, stream>>>(out, hg3, dinv, esrc, edst, E, COUT, CPAD);

  // ---- log-softmax, one wave per node ----
  k_log_softmax40<<<(N + 7) / 8, blk, 0, stream>>>(out, N);
}
